// EquiformerRegressor_50740743635146
// MI455X (gfx1250) — matrix operand order repaired and checked by reading, not compiled, not run on hardware
//
#include <hip/hip_runtime.h>

// ---------------- problem constants ----------------
#define Nn   10000
#define Ee   320000
#define Gg   32
#define Ll   6
#define C0   128
#define C1   64
#define C2   32
#define Hh   4
#define NB   128
#define RAD  64
#define Ff   512
#define MAXR 5.0f

typedef __attribute__((ext_vector_type(16))) __bf16 v16bf;
typedef __attribute__((ext_vector_type(8)))  __bf16 v8bf;
typedef __attribute__((ext_vector_type(8)))  float  v8f;
typedef __attribute__((ext_vector_type(4)))  float  v4f;

#define WMMA_BF16(a, b, c) \
  __builtin_amdgcn_wmma_f32_16x16x32_bf16(false, (a), false, (b), (short)0, (c), false, false)

static __device__ __forceinline__ v8f zero8() {
  v8f z = {0.f, 0.f, 0.f, 0.f, 0.f, 0.f, 0.f, 0.f};
  return z;
}

// A fragment: 16x32 bf16 row-major; caller passes this lane's row pointer at
// the current K offset (16B aligned). Lanes 0-15 hold K {0..7,16..23} of row
// (lane&15); lanes 16-31 hold K {8..15,24..31}.  Two b128 loads.
static __device__ __forceinline__ v16bf a_frag_bf(const __bf16* row) {
  const int lane = threadIdx.x & 31;
  const int kb = (lane & 16) ? 8 : 0;
  v8bf lo = *(const v8bf*)(row + kb);
  v8bf hi = *(const v8bf*)(row + 16 + kb);
  return __builtin_shufflevector(lo, hi, 0, 1, 2, 3, 4, 5, 6, 7,
                                 8, 9, 10, 11, 12, 13, 14, 15);
}
// Same, but f32 source (converted on the fly): four b128 loads + pack-cvt.
static __device__ __forceinline__ v16bf a_frag_f32(const float* row) {
  const int lane = threadIdx.x & 31;
  const int kb = (lane & 16) ? 8 : 0;
  v4f f0 = *(const v4f*)(row + kb);
  v4f f1 = *(const v4f*)(row + kb + 4);
  v4f f2 = *(const v4f*)(row + 16 + kb);
  v4f f3 = *(const v4f*)(row + 16 + kb + 4);
  v16bf a;
#pragma unroll
  for (int i = 0; i < 4; ++i) {
    a[i] = (__bf16)f0[i]; a[4 + i] = (__bf16)f1[i];
    a[8 + i] = (__bf16)f2[i]; a[12 + i] = (__bf16)f3[i];
  }
  return a;
}
// B fragment from pre-packed fragment-major bf16 weights: lane's 16 values are
// 32 contiguous bytes -> two b128 loads, zero converts.
static __device__ __forceinline__ v16bf b_frag_pk(const __bf16* pk, int tile) {
  const int lane = threadIdx.x & 31;
  return *(const v16bf*)(pk + (((size_t)tile << 5) + lane) * 16);
}

static __device__ __forceinline__ float siluf(float x) { return x / (1.f + __expf(-x)); }
static __device__ __forceinline__ float geluf(float x) {
  return 0.5f * x * (1.f + tanhf(0.7978845608028654f * (x + 0.044715f * x * x * x)));
}
static __device__ __forceinline__ unsigned fenc(float x) {
  unsigned b = __float_as_uint(x);
  return (b & 0x80000000u) ? ~b : (b | 0x80000000u);
}
static __device__ __forceinline__ float fdec(unsigned e) {
  unsigned b = (e & 0x80000000u) ? (e & 0x7fffffffu) : ~e;
  return __uint_as_float(b);
}

// packed-weight region offsets (elements) within one layer's block
#define PK_Wrad1   0
#define PK_Wrad2   (PK_Wrad1 + NB * RAD)
#define PK_Ws_src  (PK_Wrad2 + RAD * RAD)
#define PK_Ww_s    (PK_Ws_src + C0 * C0)
#define PK_Ws_v    (PK_Ww_s + RAD * C0)
#define PK_Ww_v    (PK_Ws_v + C0 * C1)
#define PK_Wv_v    (PK_Ww_v + RAD * C1)
#define PK_Ww_vv   (PK_Wv_v + C1 * C1)
#define PK_Ws_t    (PK_Ww_vv + RAD * C1)
#define PK_Ww_t    (PK_Ws_t + C0 * C2)
#define PK_Wt_t    (PK_Ww_t + RAD * C2)
#define PK_Ww_tt   (PK_Wt_t + C2 * C2)
#define PK_Wo_s    (PK_Ww_tt + RAD * C2)
#define PK_Wo_v    (PK_Wo_s + C0 * C0)
#define PK_Wo_t    (PK_Wo_v + C1 * C1)
#define PK_LAYER   (PK_Wo_t + C2 * C2)

struct IN {
  const float* pos; const int* node_atom; const int* batch; const int* esrc; const int* edst;
  const float* atom_emb;
  const float *Wrad1, *brad1, *Wrad2, *brad2;
  const float *Ws_src, *Ww_s, *Ws_v, *Ww_v, *Wv_v, *Ww_vv, *Ws_t, *Ww_t, *Wt_t, *Ww_tt;
  const float *attn_a, *Wo_s, *Wo_v, *Wo_t, *g_s, *b_s, *g_v, *g_t;
  const float *W_feat, *b_feat, *W_out1, *b_out1, *W_read, *b_read;
};
struct WS {
  float *s, *v, *t;            // v,t comp-major: [comp][node][ch]
  __bf16 *s_bf, *v_bf, *t_bf;  // bf16 mirrors, same layouts
  float *d_e, *u_e;
  __bf16 *w_bf, *msg_s;
  float *logit, *exv, *den;
  unsigned *mxu;
  float *agg_s, *agg_v, *agg_t; // agg_v/agg_t comp-major
  float *graph;
  __bf16 *pk, *pkW;            // packed weights (per-layer blocks), packed W_feat
};

// ---------------- weight packing: fragment-major bf16 ----------------
struct PkReg { const float* s; __bf16* d; int K, N; };

__global__ void __launch_bounds__(256) k_pack_layer(IN in, WS ws, int l) {
  int idx = blockIdx.x * 256 + threadIdx.x;
  __bf16* base = ws.pk + (size_t)l * PK_LAYER;
  PkReg regs[15] = {
    { in.Wrad1  + (size_t)l * NB * RAD,  base + PK_Wrad1,  NB,  RAD },
    { in.Wrad2  + (size_t)l * RAD * RAD, base + PK_Wrad2,  RAD, RAD },
    { in.Ws_src + (size_t)l * C0 * C0,   base + PK_Ws_src, C0,  C0 },
    { in.Ww_s   + (size_t)l * RAD * C0,  base + PK_Ww_s,   RAD, C0 },
    { in.Ws_v   + (size_t)l * C0 * C1,   base + PK_Ws_v,   C0,  C1 },
    { in.Ww_v   + (size_t)l * RAD * C1,  base + PK_Ww_v,   RAD, C1 },
    { in.Wv_v   + (size_t)l * C1 * C1,   base + PK_Wv_v,   C1,  C1 },
    { in.Ww_vv  + (size_t)l * RAD * C1,  base + PK_Ww_vv,  RAD, C1 },
    { in.Ws_t   + (size_t)l * C0 * C2,   base + PK_Ws_t,   C0,  C2 },
    { in.Ww_t   + (size_t)l * RAD * C2,  base + PK_Ww_t,   RAD, C2 },
    { in.Wt_t   + (size_t)l * C2 * C2,   base + PK_Wt_t,   C2,  C2 },
    { in.Ww_tt  + (size_t)l * RAD * C2,  base + PK_Ww_tt,  RAD, C2 },
    { in.Wo_s   + (size_t)l * C0 * C0,   base + PK_Wo_s,   C0,  C0 },
    { in.Wo_v   + (size_t)l * C1 * C1,   base + PK_Wo_v,   C1,  C1 },
    { in.Wo_t   + (size_t)l * C2 * C2,   base + PK_Wo_t,   C2,  C2 },
  };
#pragma unroll 1
  for (int m = 0; m < 15; ++m) {
    int sz = regs[m].K * regs[m].N;
    if (idx < sz) {
      int tile = idx >> 9, rem = idx & 511;
      int lane = rem >> 4, i = rem & 15;
      int nt = regs[m].N >> 4;
      int ks = tile / nt, ct = tile % nt;
      int k = ks * 32 + ((lane & 16) ? 8 : 0) + (i & 7) + ((i & 8) ? 16 : 0);
      int n = ct * 16 + (lane & 15);
      regs[m].d[idx] = (__bf16)regs[m].s[(size_t)k * regs[m].N + n];
      return;
    }
    idx -= sz;
  }
}

__global__ void __launch_bounds__(256) k_pack_one(const float* src, __bf16* dst, int K, int N) {
  int idx = blockIdx.x * 256 + threadIdx.x;
  if (idx >= K * N) return;
  int tile = idx >> 9, rem = idx & 511;
  int lane = rem >> 4, i = rem & 15;
  int nt = N >> 4;
  int ks = tile / nt, ct = tile % nt;
  int k = ks * 32 + ((lane & 16) ? 8 : 0) + (i & 7) + ((i & 8) ? 16 : 0);
  int n = ct * 16 + (lane & 15);
  dst[idx] = (__bf16)src[(size_t)k * N + n];
}

// ---------------- setup kernels ----------------
__global__ void __launch_bounds__(256) k_init(IN in, WS ws) {
  size_t i = (size_t)blockIdx.x * 256 + threadIdx.x;
  if (i < (size_t)Nn * C0) {
    int node = (int)(i / C0), c = (int)(i % C0);
    float val = in.atom_emb[in.node_atom[node] * C0 + c];
    ws.s[i] = val; ws.s_bf[i] = (__bf16)val;
  }
  if (i < (size_t)Nn * C1 * 3) { ws.v[i] = 0.f; ws.v_bf[i] = (__bf16)0.f; }
  if (i < (size_t)Nn * C2 * 5) { ws.t[i] = 0.f; ws.t_bf[i] = (__bf16)0.f; }
  if (i < Gg) ws.graph[i] = 0.f;
}

__global__ void __launch_bounds__(256) k_geom(IN in, WS ws) {
  int e = blockIdx.x * 256 + threadIdx.x;
  if (e >= Ee) return;
  int sI = in.esrc[e], dI = in.edst[e];
  float rx = in.pos[dI * 3 + 0] - in.pos[sI * 3 + 0];
  float ry = in.pos[dI * 3 + 1] - in.pos[sI * 3 + 1];
  float rz = in.pos[dI * 3 + 2] - in.pos[sI * 3 + 2];
  float dd = sqrtf(rx * rx + ry * ry + rz * rz + 1e-6f);
  ws.d_e[e] = dd;
  ws.u_e[(size_t)e * 3 + 0] = rx / dd;
  ws.u_e[(size_t)e * 3 + 1] = ry / dd;
  ws.u_e[(size_t)e * 3 + 2] = rz / dd;
}

__global__ void __launch_bounds__(256) k_zero_layer(WS ws) {
  size_t i = (size_t)blockIdx.x * 256 + threadIdx.x;
  if (i < (size_t)Nn * C0) ws.agg_s[i] = 0.f;
  if (i < (size_t)Nn * C1 * 3) ws.agg_v[i] = 0.f;
  if (i < (size_t)Nn * C2 * 5) ws.agg_t[i] = 0.f;
  if (i < (size_t)Nn * Hh) { ws.mxu[i] = 0u; ws.den[i] = 1e-9f; }
}

// ---------------- radial MLP: w = silu(silu(rbf@W1+b1)@W2+b2) ----------------
__global__ void __launch_bounds__(256) k_radial(IN in, WS ws, int l) {
  const int wave = threadIdx.x >> 5, lane = threadIdx.x & 31;
  const int e0 = (blockIdx.x * 8 + wave) * 16;
  __shared__ __align__(32) __bf16 rbfT[8][16 * NB];
  __shared__ __align__(32) __bf16 h1T[8][16 * RAD];
  const float width = MAXR / NB;
  const float cstep = MAXR / (NB - 1);
  for (int idx = lane; idx < 16 * NB; idx += 32) {
    int er = idx >> 7, k = idx & (NB - 1);
    float dd = ws.d_e[e0 + er];
    float z = (dd - cstep * (float)k) / width;
    rbfT[wave][idx] = (__bf16)__expf(-0.5f * z * z);
  }
  __syncthreads();
  const __bf16* pW1 = ws.pk + (size_t)l * PK_LAYER + PK_Wrad1;
  const __bf16* pW2 = ws.pk + (size_t)l * PK_LAYER + PK_Wrad2;
  const int r = lane & 15, n = r, mo = (lane >> 4) * 8;
#pragma unroll
  for (int ct = 0; ct < 4; ++ct) {
    v8f acc = zero8();
#pragma unroll
    for (int ks = 0; ks < 4; ++ks)
      acc = WMMA_BF16(a_frag_bf(&rbfT[wave][r * NB + ks * 32]), b_frag_pk(pW1, ks * 4 + ct), acc);
    float bias = in.brad1[l * RAD + ct * 16 + n];
#pragma unroll
    for (int rr = 0; rr < 8; ++rr) {
      float x = acc[rr] + bias;
      h1T[wave][(mo + rr) * RAD + ct * 16 + n] = (__bf16)siluf(x);
    }
  }
  __syncthreads();
#pragma unroll
  for (int ct = 0; ct < 4; ++ct) {
    v8f acc = zero8();
#pragma unroll
    for (int ks = 0; ks < 2; ++ks)
      acc = WMMA_BF16(a_frag_bf(&h1T[wave][r * RAD + ks * 32]), b_frag_pk(pW2, ks * 4 + ct), acc);
    float bias = in.brad2[l * RAD + ct * 16 + n];
#pragma unroll
    for (int rr = 0; rr < 8; ++rr) {
      float x = acc[rr] + bias;
      ws.w_bf[(size_t)(e0 + mo + rr) * RAD + ct * 16 + n] = (__bf16)siluf(x);
    }
  }
}

// ---------------- scalar messages: msg_s = (s_e@Ws_src) * (w@Ww_s) ----------------
__global__ void __launch_bounds__(256) k_msg_s(IN in, WS ws, int l) {
  const int wave = threadIdx.x >> 5, lane = threadIdx.x & 31;
  const int e0 = (blockIdx.x * 8 + wave) * 16;
  const int r = lane & 15, n = r, mo = (lane >> 4) * 8;
  const int src = in.esrc[e0 + r];
  const __bf16* srow = ws.s_bf + (size_t)src * C0;
  const __bf16* wrow = ws.w_bf + (size_t)(e0 + r) * RAD;
  const __bf16* pWss = ws.pk + (size_t)l * PK_LAYER + PK_Ws_src;
  const __bf16* pWws = ws.pk + (size_t)l * PK_LAYER + PK_Ww_s;
#pragma unroll
  for (int ct = 0; ct < 8; ++ct) {
    v8f a1 = zero8();
#pragma unroll
    for (int ks = 0; ks < 4; ++ks)
      a1 = WMMA_BF16(a_frag_bf(srow + ks * 32), b_frag_pk(pWss, ks * 8 + ct), a1);
    v8f a2 = zero8();
#pragma unroll
    for (int ks = 0; ks < 2; ++ks)
      a2 = WMMA_BF16(a_frag_bf(wrow + ks * 32), b_frag_pk(pWws, ks * 8 + ct), a2);
#pragma unroll
    for (int rr = 0; rr < 8; ++rr)
      ws.msg_s[(size_t)(e0 + mo + rr) * C0 + ct * 16 + n] = (__bf16)(a1[rr] * a2[rr]);
  }
}

// ---------------- attention logits / segment softmax ----------------
__global__ void __launch_bounds__(256) k_logit(IN in, WS ws, int l) {
  size_t i = (size_t)blockIdx.x * 256 + threadIdx.x;
  if (i >= (size_t)Ee * Hh) return;
  int e = (int)(i >> 2), h = (int)(i & 3);
  const __bf16* m = ws.msg_s + (size_t)e * C0 + h * 32;
  const float* a = in.attn_a + (size_t)(l * Hh + h) * (C0 / Hh);
  float s = 0.f;
#pragma unroll
  for (int c = 0; c < 32; ++c) s += (float)m[c] * a[c];
  ws.logit[i] = s;
}

__global__ void __launch_bounds__(256) k_segmax(IN in, WS ws) {
  size_t i = (size_t)blockIdx.x * 256 + threadIdx.x;
  if (i >= (size_t)Ee * Hh) return;
  int e = (int)(i >> 2), h = (int)(i & 3);
  atomicMax(&ws.mxu[(size_t)in.edst[e] * Hh + h], fenc(ws.logit[i]));
}

__global__ void __launch_bounds__(256) k_segexp(IN in, WS ws) {
  size_t i = (size_t)blockIdx.x * 256 + threadIdx.x;
  if (i >= (size_t)Ee * Hh) return;
  int e = (int)(i >> 2), h = (int)(i & 3);
  int dd = in.edst[e];
  float ex = __expf(ws.logit[i] - fdec(ws.mxu[(size_t)dd * Hh + h]));
  ws.exv[i] = ex;
  atomicAdd(&ws.den[(size_t)dd * Hh + h], ex);
}

__global__ void __launch_bounds__(256) k_agg_s(IN in, WS ws) {
  size_t i = (size_t)blockIdx.x * 256 + threadIdx.x;
  if (i >= (size_t)Ee * C0) return;
  int e = (int)(i >> 7), c = (int)(i & 127), h = c >> 5;
  int dd = in.edst[e];
  float al = ws.exv[(size_t)e * Hh + h] / ws.den[(size_t)dd * Hh + h];
  atomicAdd(&ws.agg_s[(size_t)dd * C0 + c], al * (float)ws.msg_s[i]);
}

// ---------------- vector/tensor messages + fused alpha-weighted scatter ----------------
__global__ void __launch_bounds__(256) k_msg_vt(IN in, WS ws, int l) {
  const int wave = threadIdx.x >> 5, lane = threadIdx.x & 31;
  const int e0 = (blockIdx.x * 8 + wave) * 16;
  __shared__ float alphaT[8][16][4];
  __shared__ float sh1T[8][16][3];
  __shared__ float sh2T[8][16][5];
  __shared__ int dstT[8][16];
  for (int idx = lane; idx < 64; idx += 32) {
    int rr = idx >> 2, h = idx & 3;
    int e = e0 + rr, dd = in.edst[e];
    alphaT[wave][rr][h] = ws.exv[(size_t)e * Hh + h] / ws.den[(size_t)dd * Hh + h];
  }
  if (lane < 16) {
    int e = e0 + lane;
    dstT[wave][lane] = in.edst[e];
    float ux = ws.u_e[(size_t)e * 3 + 0], uy = ws.u_e[(size_t)e * 3 + 1], uz = ws.u_e[(size_t)e * 3 + 2];
    const float s3 = 1.7320508075688772f, s5 = 2.23606797749979f, s15 = 3.872983346207417f;
    sh1T[wave][lane][0] = s3 * ux; sh1T[wave][lane][1] = s3 * uy; sh1T[wave][lane][2] = s3 * uz;
    sh2T[wave][lane][0] = s15 * ux * uy;
    sh2T[wave][lane][1] = s15 * uy * uz;
    sh2T[wave][lane][2] = 0.5f * s5 * (3.f * uz * uz - 1.f);
    sh2T[wave][lane][3] = s15 * ux * uz;
    sh2T[wave][lane][4] = 0.5f * s15 * (ux * ux - uy * uy);
  }
  __syncthreads();
  const int r = lane & 15, n = r, mo = (lane >> 4) * 8;
  const int src = in.esrc[e0 + r];
  const __bf16* srow = ws.s_bf + (size_t)src * C0;
  const __bf16* wrow = ws.w_bf + (size_t)(e0 + r) * RAD;
  const __bf16* base = ws.pk + (size_t)l * PK_LAYER;
  const __bf16 *pWsv = base + PK_Ws_v, *pWwv = base + PK_Ww_v;
  const __bf16 *pWvv = base + PK_Wv_v, *pWwvv = base + PK_Ww_vv;
  const __bf16 *pWst = base + PK_Ws_t, *pWwt = base + PK_Ww_t;
  const __bf16 *pWtt = base + PK_Wt_t, *pWwtt = base + PK_Ww_tt;

  // vector channels (C1=64 -> 4 column tiles; head = ct since C1/H = 16)
#pragma unroll
  for (int ct = 0; ct < 4; ++ct) {
    v8f gv = zero8();
#pragma unroll
    for (int ks = 0; ks < 4; ++ks)
      gv = WMMA_BF16(a_frag_bf(srow + ks * 32), b_frag_pk(pWsv, ks * 4 + ct), gv);
    v8f gw = zero8();
#pragma unroll
    for (int ks = 0; ks < 2; ++ks)
      gw = WMMA_BF16(a_frag_bf(wrow + ks * 32), b_frag_pk(pWwv, ks * 4 + ct), gw);
    v8f wvv = zero8();
#pragma unroll
    for (int ks = 0; ks < 2; ++ks)
      wvv = WMMA_BF16(a_frag_bf(wrow + ks * 32), b_frag_pk(pWwvv, ks * 4 + ct), wvv);
#pragma unroll
    for (int comp = 0; comp < 3; ++comp) {
      const __bf16* vrow = ws.v_bf + (size_t)comp * Nn * C1 + (size_t)src * C1;
      v8f vr = zero8();
#pragma unroll
      for (int ks = 0; ks < 2; ++ks)
        vr = WMMA_BF16(a_frag_bf(vrow + ks * 32), b_frag_pk(pWvv, ks * 4 + ct), vr);
#pragma unroll
      for (int rr = 0; rr < 8; ++rr) {
        int row = mo + rr;
        float msg = gv[rr] * gw[rr] * sh1T[wave][row][comp] + vr[rr] * wvv[rr];
        float al = alphaT[wave][row][ct];
        atomicAdd(&ws.agg_v[(size_t)comp * Nn * C1 + (size_t)dstT[wave][row] * C1 + ct * 16 + n],
                  al * msg);
      }
    }
  }
  // tensor channels (C2=32 -> 2 column tiles; head = (ct*16+n)>>3 since C2/H = 8)
#pragma unroll
  for (int ct = 0; ct < 2; ++ct) {
    v8f gt = zero8();
#pragma unroll
    for (int ks = 0; ks < 4; ++ks)
      gt = WMMA_BF16(a_frag_bf(srow + ks * 32), b_frag_pk(pWst, ks * 2 + ct), gt);
    v8f gw = zero8();
#pragma unroll
    for (int ks = 0; ks < 2; ++ks)
      gw = WMMA_BF16(a_frag_bf(wrow + ks * 32), b_frag_pk(pWwt, ks * 2 + ct), gw);
    v8f wtt = zero8();
#pragma unroll
    for (int ks = 0; ks < 2; ++ks)
      wtt = WMMA_BF16(a_frag_bf(wrow + ks * 32), b_frag_pk(pWwtt, ks * 2 + ct), wtt);
    const int h = (ct * 16 + n) >> 3;
#pragma unroll
    for (int comp = 0; comp < 5; ++comp) {
      const __bf16* trow = ws.t_bf + (size_t)comp * Nn * C2 + (size_t)src * C2;
      v8f tr = zero8();
      tr = WMMA_BF16(a_frag_bf(trow), b_frag_pk(pWtt, ct), tr);
#pragma unroll
      for (int rr = 0; rr < 8; ++rr) {
        int row = mo + rr;
        float msg = gt[rr] * gw[rr] * sh2T[wave][row][comp] + tr[rr] * wtt[rr];
        float al = alphaT[wave][row][h];
        atomicAdd(&ws.agg_t[(size_t)comp * Nn * C2 + (size_t)dstT[wave][row] * C2 + ct * 16 + n],
                  al * msg);
      }
    }
  }
}

// ---------------- node update: residual GEMMs + norms (all in registers) ----------------
__global__ void __launch_bounds__(160) k_node_update(IN in, WS ws, int l) {
  const int wave = threadIdx.x >> 5, lane = threadIdx.x & 31;
  const int n0 = (blockIdx.x * 5 + wave) * 16;
  const int r = lane & 15, n = r, mo = (lane >> 4) * 8;
  const float eps = 1e-6f;
  const __bf16* base = ws.pk + (size_t)l * PK_LAYER;
  // ---- scalars: s += agg_s @ Wo_s; layernorm ----
  {
    const __bf16* pWos = base + PK_Wo_s;
    v8f cs[8];
    float sum[8], sq[8];
#pragma unroll
    for (int rr = 0; rr < 8; ++rr) { sum[rr] = 0.f; sq[rr] = 0.f; }
#pragma unroll
    for (int ct = 0; ct < 8; ++ct) {
      v8f acc = zero8();
#pragma unroll
      for (int ks = 0; ks < 4; ++ks)
        acc = WMMA_BF16(a_frag_f32(ws.agg_s + (size_t)(n0 + r) * C0 + ks * 32),
                        b_frag_pk(pWos, ks * 8 + ct), acc);
#pragma unroll
      for (int rr = 0; rr < 8; ++rr) {
        float x = ws.s[(size_t)(n0 + mo + rr) * C0 + ct * 16 + n] + acc[rr];
        cs[ct][rr] = x; sum[rr] += x; sq[rr] += x * x;
      }
    }
#pragma unroll
    for (int m = 1; m < 16; m <<= 1)
#pragma unroll
      for (int rr = 0; rr < 8; ++rr) {
        sum[rr] += __shfl_xor(sum[rr], m, 32);
        sq[rr] += __shfl_xor(sq[rr], m, 32);
      }
#pragma unroll
    for (int ct = 0; ct < 8; ++ct)
#pragma unroll
      for (int rr = 0; rr < 8; ++rr) {
        float mu = sum[rr] * (1.f / C0);
        float var = sq[rr] * (1.f / C0) - mu * mu;
        float inv = rsqrtf(var + eps);
        int ch = ct * 16 + n;
        float val = (cs[ct][rr] - mu) * inv * in.g_s[l * C0 + ch] + in.b_s[l * C0 + ch];
        size_t ix = (size_t)(n0 + mo + rr) * C0 + ch;
        ws.s[ix] = val; ws.s_bf[ix] = (__bf16)val;
      }
  }
  // ---- vectors: v += agg_v @ Wo_v; equivariant norm ----
  {
    const __bf16* pWov = base + PK_Wo_v;
    v8f cv[3][4];
    float sq[8];
#pragma unroll
    for (int rr = 0; rr < 8; ++rr) sq[rr] = 0.f;
#pragma unroll
    for (int comp = 0; comp < 3; ++comp)
#pragma unroll
      for (int ct = 0; ct < 4; ++ct) {
        v8f acc = zero8();
#pragma unroll
        for (int ks = 0; ks < 2; ++ks)
          acc = WMMA_BF16(a_frag_f32(ws.agg_v + (size_t)comp * Nn * C1 + (size_t)(n0 + r) * C1 + ks * 32),
                          b_frag_pk(pWov, ks * 4 + ct), acc);
#pragma unroll
        for (int rr = 0; rr < 8; ++rr) {
          float x = ws.v[(size_t)comp * Nn * C1 + (size_t)(n0 + mo + rr) * C1 + ct * 16 + n] + acc[rr];
          cv[comp][ct][rr] = x; sq[rr] += x * x;
        }
      }
#pragma unroll
    for (int m = 1; m < 16; m <<= 1)
#pragma unroll
      for (int rr = 0; rr < 8; ++rr) sq[rr] += __shfl_xor(sq[rr], m, 32);
#pragma unroll
    for (int comp = 0; comp < 3; ++comp)
#pragma unroll
      for (int ct = 0; ct < 4; ++ct)
#pragma unroll
        for (int rr = 0; rr < 8; ++rr) {
          float inv = rsqrtf(sq[rr] * (1.f / C1) + eps);
          int ch = ct * 16 + n;
          float val = cv[comp][ct][rr] * inv * in.g_v[l * C1 + ch];
          size_t ix = (size_t)comp * Nn * C1 + (size_t)(n0 + mo + rr) * C1 + ch;
          ws.v[ix] = val; ws.v_bf[ix] = (__bf16)val;
        }
  }
  // ---- tensors: t += agg_t @ Wo_t; equivariant norm ----
  {
    const __bf16* pWot = base + PK_Wo_t;
    v8f ctf[5][2];
    float sq[8];
#pragma unroll
    for (int rr = 0; rr < 8; ++rr) sq[rr] = 0.f;
#pragma unroll
    for (int comp = 0; comp < 5; ++comp)
#pragma unroll
      for (int ct = 0; ct < 2; ++ct) {
        v8f acc = zero8();
        acc = WMMA_BF16(a_frag_f32(ws.agg_t + (size_t)comp * Nn * C2 + (size_t)(n0 + r) * C2),
                        b_frag_pk(pWot, ct), acc);
#pragma unroll
        for (int rr = 0; rr < 8; ++rr) {
          float x = ws.t[(size_t)comp * Nn * C2 + (size_t)(n0 + mo + rr) * C2 + ct * 16 + n] + acc[rr];
          ctf[comp][ct][rr] = x; sq[rr] += x * x;
        }
      }
#pragma unroll
    for (int m = 1; m < 16; m <<= 1)
#pragma unroll
      for (int rr = 0; rr < 8; ++rr) sq[rr] += __shfl_xor(sq[rr], m, 32);
#pragma unroll
    for (int comp = 0; comp < 5; ++comp)
#pragma unroll
      for (int ct = 0; ct < 2; ++ct)
#pragma unroll
        for (int rr = 0; rr < 8; ++rr) {
          float inv = rsqrtf(sq[rr] * (1.f / C2) + eps);
          int ch = ct * 16 + n;
          float val = ctf[comp][ct][rr] * inv * in.g_t[l * C2 + ch];
          size_t ix = (size_t)comp * Nn * C2 + (size_t)(n0 + mo + rr) * C2 + ch;
          ws.t[ix] = val; ws.t_bf[ix] = (__bf16)val;
        }
  }
}

// ---------------- readout: gelu(s@W_feat+b)@W_out1 + b_out1, per-graph sum ----------------
__global__ void __launch_bounds__(160) k_readout(IN in, WS ws) {
  const int wave = threadIdx.x >> 5, lane = threadIdx.x & 31;
  const int n0 = (blockIdx.x * 5 + wave) * 16;
  const int r = lane & 15, n = r, mo = (lane >> 4) * 8;
  const __bf16* srow = ws.s_bf + (size_t)(n0 + r) * C0;
  float pe[8];
#pragma unroll
  for (int rr = 0; rr < 8; ++rr) pe[rr] = 0.f;
  for (int ct = 0; ct < Ff / 16; ++ct) {
    v8f acc = zero8();
#pragma unroll
    for (int ks = 0; ks < 4; ++ks)
      acc = WMMA_BF16(a_frag_bf(srow + ks * 32), b_frag_pk(ws.pkW, ks * 32 + ct), acc);
    float bf = in.b_feat[ct * 16 + n];
    float wo = in.W_out1[ct * 16 + n];
#pragma unroll
    for (int rr = 0; rr < 8; ++rr) pe[rr] += geluf(acc[rr] + bf) * wo;
  }
#pragma unroll
  for (int m = 1; m < 16; m <<= 1)
#pragma unroll
    for (int rr = 0; rr < 8; ++rr) pe[rr] += __shfl_xor(pe[rr], m, 32);
  if ((lane & 15) == 0) {
#pragma unroll
    for (int rr = 0; rr < 8; ++rr) {
      int node = n0 + mo + rr;
      atomicAdd(&ws.graph[in.batch[node]], pe[rr] + in.b_out1[0]);
    }
  }
}

__global__ void k_final(IN in, WS ws, float* out) {
  int g = threadIdx.x;
  if (g < Gg) out[g] = ws.graph[g] * in.W_read[0] + in.b_read[0];
}

// ---------------- host ----------------
extern "C" void kernel_launch(void* const* d_in, const int* in_sizes, int n_in,
                              void* d_out, int out_size, void* d_ws, size_t ws_size,
                              hipStream_t stream) {
  (void)in_sizes; (void)n_in; (void)out_size; (void)ws_size;
  IN in;
  in.pos = (const float*)d_in[0];
  in.node_atom = (const int*)d_in[1];
  in.batch = (const int*)d_in[2];
  in.esrc = (const int*)d_in[3];
  in.edst = (const int*)d_in[4];
  in.atom_emb = (const float*)d_in[5];
  in.Wrad1 = (const float*)d_in[6];  in.brad1 = (const float*)d_in[7];
  in.Wrad2 = (const float*)d_in[8];  in.brad2 = (const float*)d_in[9];
  in.Ws_src = (const float*)d_in[10]; in.Ww_s = (const float*)d_in[11];
  in.Ws_v = (const float*)d_in[12];  in.Ww_v = (const float*)d_in[13];
  in.Wv_v = (const float*)d_in[14];  in.Ww_vv = (const float*)d_in[15];
  in.Ws_t = (const float*)d_in[16];  in.Ww_t = (const float*)d_in[17];
  in.Wt_t = (const float*)d_in[18];  in.Ww_tt = (const float*)d_in[19];
  in.attn_a = (const float*)d_in[20];
  in.Wo_s = (const float*)d_in[21];  in.Wo_v = (const float*)d_in[22]; in.Wo_t = (const float*)d_in[23];
  in.g_s = (const float*)d_in[24];   in.b_s = (const float*)d_in[25];
  in.g_v = (const float*)d_in[26];   in.g_t = (const float*)d_in[27];
  in.W_feat = (const float*)d_in[28]; in.b_feat = (const float*)d_in[29];
  in.W_out1 = (const float*)d_in[30]; in.b_out1 = (const float*)d_in[31];
  in.W_read = (const float*)d_in[32]; in.b_read = (const float*)d_in[33];

  char* p = (char*)d_ws;
  auto take = [&](size_t bytes) -> char* {
    char* q = p;
    p += (bytes + 255) & ~(size_t)255;
    return q;
  };
  WS ws;
  ws.s     = (float*)take((size_t)Nn * C0 * 4);
  ws.v     = (float*)take((size_t)Nn * C1 * 3 * 4);
  ws.t     = (float*)take((size_t)Nn * C2 * 5 * 4);
  ws.s_bf  = (__bf16*)take((size_t)Nn * C0 * 2);
  ws.v_bf  = (__bf16*)take((size_t)Nn * C1 * 3 * 2);
  ws.t_bf  = (__bf16*)take((size_t)Nn * C2 * 5 * 2);
  ws.d_e   = (float*)take((size_t)Ee * 4);
  ws.u_e   = (float*)take((size_t)Ee * 3 * 4);
  ws.w_bf  = (__bf16*)take((size_t)Ee * RAD * 2);
  ws.msg_s = (__bf16*)take((size_t)Ee * C0 * 2);
  ws.logit = (float*)take((size_t)Ee * Hh * 4);
  ws.exv   = (float*)take((size_t)Ee * Hh * 4);
  ws.den   = (float*)take((size_t)Nn * Hh * 4);
  ws.mxu   = (unsigned*)take((size_t)Nn * Hh * 4);
  ws.agg_s = (float*)take((size_t)Nn * C0 * 4);
  ws.agg_v = (float*)take((size_t)Nn * C1 * 3 * 4);
  ws.agg_t = (float*)take((size_t)Nn * C2 * 5 * 4);
  ws.graph = (float*)take((size_t)Gg * 4);
  ws.pk    = (__bf16*)take((size_t)PK_LAYER * Ll * 2);
  ws.pkW   = (__bf16*)take((size_t)C0 * Ff * 2);

  const int ETILE_BLOCKS = (Ee / 16) / 8;       // 2500 (8 wave-tiles of 16 edges / block)
  const int NTILE_BLOCKS = (Nn / 16) / 5;       // 125  (5 wave-tiles of 16 nodes / block)
  const int EH_BLOCKS = (Ee * Hh + 255) / 256;  // 5000
  const int NC_BLOCKS = (int)(((size_t)Nn * C0 + 255) / 256);  // 5000
  const int EC_BLOCKS = (int)(((size_t)Ee * C0 + 255) / 256);  // 160000

  // pack all weights into bf16 fragment-major layout (cheap, L2-resident)
  for (int l = 0; l < Ll; ++l)
    k_pack_layer<<<(PK_LAYER + 255) / 256, 256, 0, stream>>>(in, ws, l);
  k_pack_one<<<(C0 * Ff + 255) / 256, 256, 0, stream>>>(in.W_feat, ws.pkW, C0, Ff);

  k_init<<<NC_BLOCKS, 256, 0, stream>>>(in, ws);
  k_geom<<<(Ee + 255) / 256, 256, 0, stream>>>(in, ws);
  for (int l = 0; l < Ll; ++l) {
    k_zero_layer<<<NC_BLOCKS, 256, 0, stream>>>(ws);
    k_radial<<<ETILE_BLOCKS, 256, 0, stream>>>(in, ws, l);
    k_msg_s<<<ETILE_BLOCKS, 256, 0, stream>>>(in, ws, l);
    k_logit<<<EH_BLOCKS, 256, 0, stream>>>(in, ws, l);
    k_segmax<<<EH_BLOCKS, 256, 0, stream>>>(in, ws);
    k_segexp<<<EH_BLOCKS, 256, 0, stream>>>(in, ws);
    k_agg_s<<<EC_BLOCKS, 256, 0, stream>>>(in, ws);
    k_msg_vt<<<ETILE_BLOCKS, 256, 0, stream>>>(in, ws, l);
    k_node_update<<<NTILE_BLOCKS, 160, 0, stream>>>(in, ws, l);
  }
  k_readout<<<NTILE_BLOCKS, 160, 0, stream>>>(in, ws);
  k_final<<<1, 32, 0, stream>>>(in, ws, (float*)d_out);
}
